// MOTSSegClsLoss0904_32555852103991
// MI455X (gfx1250) — matrix operand order repaired and checked by reading, not compiled, not run on hardware
//
#include <hip/hip_runtime.h>
#include <hip/hip_bf16.h>
#include <math.h>

// ---------------------------------------------------------------------------
// MOTS seg+cls loss for MI455X (gfx1250, wave32).
//  - Per-id statistics via v_wmma_f32_16x16x32_f16 (onehot-mask GEMM).
//  - Lovasz hinge via descending LSD radix sort (stable wave32 ballot ranking).
//  - Workspace requirement: ~119 MB (2x key buffers 58.7MB + 7.3MB hist + small).
// ---------------------------------------------------------------------------

typedef __attribute__((ext_vector_type(16))) _Float16 v16h;
typedef __attribute__((ext_vector_type(8)))  float    v8f;

#define NPIX    524288   // 512*1024
#define NB      4
#define NID     7        // ids 1..7
#define NSEG    28       // NB*NID
#define NCHUNK  256      // sort chunks per segment (2048 elems per wave-chunk)
#define SORTC   2048
#define LCHUNK  4096     // lovasz cumsum chunk
#define NLCHUNK 128

// accumulator block layout (floats), zeroed each launch
#define STATS_OFF 0      // NSEG*5 : cnt, sum_x, sum_y, sum_s0, sum_s1
#define VAR_OFF   140    // NSEG
#define SFG_OFF   168    // NSEG
#define INSTL_OFF 196    // NSEG
#define BST_OFF   224    // NB*3 : focal_sum, valid_cnt, seed_bg
#define DER_OFF   236    // NSEG*8 : cx, cy, m0, m1, e0, e1, present, cntf
#define ACC_TOTAL 512

// ---------------------------------------------------------------------------
__global__ __launch_bounds__(256) void k0_zero(float* acc) {
  int t = threadIdx.x;
  acc[t] = 0.f;
  acc[t + 256] = 0.f;
}

// ---------------------------------------------------------------------------
// Pass 1: per-id stats via WMMA + per-batch focal / valid / seed_bg scalars.
// A (16x32 f16): rows = features {1, x, y, s0, s1}; B (32x16 f16): onehot(id).
// D[m][n] += sum_k feat_m(pix k) * (inst(pix k)==n)  -> per-id feature sums.
// ---------------------------------------------------------------------------
__global__ __launch_bounds__(256)
void k1_stats(const float* __restrict__ pred, const int* __restrict__ inst,
              const int* __restrict__ lab, float* __restrict__ stats,
              float* __restrict__ bstats) {
  __shared__ unsigned short s_inst[8][32];
  __shared__ _Float16 s_feat[8][4][32];
  __shared__ float red[256];
  const int tid = threadIdx.x, w = tid >> 5, lane = tid & 31;
  const int b = blockIdx.y;
  const int span = blockIdx.x * 8192;
  const float* pb = pred + (size_t)b * 7 * NPIX;
  const int* ib = inst + (size_t)b * NPIX;
  const int* lb_ = lab + (size_t)b * NPIX;

  v8f c = {};
  float facc = 0.f, vacc = 0.f, sbacc = 0.f;

  for (int it = 0; it < 32; ++it) {
    int p = span + it * 256 + w * 32 + lane;
    int hrow = p >> 10, wcol = p & 1023;
    float x = 2.0f * (float)wcol / 2047.0f;   // linspace(0,2,2048)[:1024]
    float y = (float)hrow / 1023.0f;          // linspace(0,1,1024)[:512]
    int iid = ib[p];
    float s0 = pb[2 * NPIX + p];
    float s1 = pb[3 * NPIX + p];
    // ---- scalar losses for this pixel ----
    int lv = lb_[p];
    float x4 = pb[4 * NPIX + p];
    float x5 = pb[5 * NPIX + p];
    float x6 = pb[6 * NPIX + p];
    float mx = fmaxf(x5, x6);
    float lse = mx + logf(expf(x5 - mx) + expf(x6 - mx));
    float logpt = ((lv > 0) ? x6 : x5) - lse;  // tgt = clip(lab,0,1)
    float pt = expf(logpt);
    float om = 1.f - pt;
    float focal = -om * om * logpt;            // GAMMA = 2
    if (lv < 2) { facc += focal; vacc += 1.f; }
    float seed = 1.f / (1.f + expf(-x4));
    if (lv == 0) sbacc += seed * seed;
    // ---- stage features for WMMA fragment build ----
    s_inst[w][lane] = (unsigned short)iid;
    s_feat[w][0][lane] = (_Float16)x;
    s_feat[w][1][lane] = (_Float16)y;
    s_feat[w][2][lane] = (_Float16)s0;
    s_feat[w][3][lane] = (_Float16)s1;
    __syncthreads();
    // A fragment: lane L holds row m=L%16; K = (h<8?h:h+8) + (L>=16?8:0)
    int m = lane & 15;
    int koff = (lane < 16) ? 0 : 8;
    v16h a, bm;
#pragma unroll
    for (int h = 0; h < 16; ++h) {
      int K = ((h < 8) ? h : h + 8) + koff;
      _Float16 v = (_Float16)0.f;
      if (m == 0) v = (_Float16)1.f;
      else if (m <= 4) v = s_feat[w][m - 1][K];
      a[h] = v;
    }
    // B fragment: lane L holds col n=L%16; K = (h<8?h:h+8) + (L>=16?8:0)
    int n = lane & 15;
#pragma unroll
    for (int h = 0; h < 16; ++h)
      bm[h] = (s_inst[w][((h < 8) ? h : h + 8) + koff] == n) ? (_Float16)1.f : (_Float16)0.f;

    c = __builtin_amdgcn_wmma_f32_16x16x32_f16(false, a, false, bm,
                                               (short)0, c, false, false);
    __syncthreads();
  }

  // D layout: lanes 0..15 -> N=lane, reg r -> M=r (features 0..4). ids 1..7.
  if (lane >= 1 && lane <= 7) {
#pragma unroll
    for (int r = 0; r < 5; ++r)
      atomicAdd(&stats[((size_t)b * NID + (lane - 1)) * 5 + r], c[r]);
  }

  // block reductions for scalar losses
  red[tid] = facc; __syncthreads();
  for (int st = 128; st > 0; st >>= 1) { if (tid < st) red[tid] += red[tid + st]; __syncthreads(); }
  if (tid == 0) atomicAdd(&bstats[b * 3 + 0], red[0]);
  __syncthreads();
  red[tid] = vacc; __syncthreads();
  for (int st = 128; st > 0; st >>= 1) { if (tid < st) red[tid] += red[tid + st]; __syncthreads(); }
  if (tid == 0) atomicAdd(&bstats[b * 3 + 1], red[0]);
  __syncthreads();
  red[tid] = sbacc; __syncthreads();
  for (int st = 128; st > 0; st >>= 1) { if (tid < st) red[tid] += red[tid + st]; __syncthreads(); }
  if (tid == 0) atomicAdd(&bstats[b * 3 + 2], red[0]);
}

// ---------------------------------------------------------------------------
__global__ void k2_centers(float* acc) {
  int s = threadIdx.x;
  if (s < NSEG) {
    const float* st = acc + STATS_OFF + s * 5;
    float cnt = st[0];
    float cntf = fmaxf(cnt, 1.f);
    float cx = st[1] / cntf, cy = st[2] / cntf;
    float m0 = st[3] / cntf, m1 = st[4] / cntf;
    float* d = acc + DER_OFF + s * 8;
    d[0] = cx; d[1] = cy; d[2] = m0; d[3] = m1;
    d[4] = expf(10.f * m0); d[5] = expf(10.f * m1);
    d[6] = (cnt > 0.f) ? 1.f : 0.f; d[7] = cntf;
  }
}

// ---------------------------------------------------------------------------
// Pass 2: per (seg=b,id) pixel map -> sort key, var_l partial, seed_fg partial.
// key = (bits(err) << 1) | mask  (err >= 0 so sign bit is 0); sorted descending.
// ---------------------------------------------------------------------------
__global__ __launch_bounds__(256)
void k3_dist(const float* __restrict__ pred, const int* __restrict__ inst,
             const float* __restrict__ derived, unsigned* __restrict__ keys,
             float* __restrict__ varsum, float* __restrict__ seedfg) {
  __shared__ float red[256];
  int seg = blockIdx.y;
  int b = seg / NID, id = seg % NID + 1;
  const float* d8 = derived + seg * 8;
  float cx = d8[0], cy = d8[1], m0 = d8[2], m1 = d8[3], e0 = d8[4], e1 = d8[5];
  const float* pb = pred + (size_t)b * 7 * NPIX;
  const int* ib = inst + (size_t)b * NPIX;
  int base = blockIdx.x * LCHUNK;
  float vacc = 0.f, sacc = 0.f;
  for (int i = 0; i < 16; ++i) {
    int p = base + i * 256 + threadIdx.x;
    int hrow = p >> 10, wcol = p & 1023;
    float x = 2.f * (float)wcol / 2047.f;
    float y = (float)hrow / 1023.f;
    float ex = tanhf(pb[p]) + x;
    float ey = tanhf(pb[NPIX + p]) + y;
    float dx = ex - cx, dy = ey - cy;
    float d = expf(-(dx * dx * e0 + dy * dy * e1));
    int mk = (ib[p] == id) ? 1 : 0;
    float err = mk ? (2.f - 2.f * d) : (2.f * d);
    keys[(size_t)seg * NPIX + p] = (__float_as_uint(err) << 1) | (unsigned)mk;
    if (mk) {
      float t0 = pb[2 * NPIX + p] - m0, t1 = pb[3 * NPIX + p] - m1;
      vacc += t0 * t0 + t1 * t1;
      float seed = 1.f / (1.f + expf(-pb[4 * NPIX + p]));
      float q = seed - d;
      sacc += q * q;
    }
  }
  int tid = threadIdx.x;
  red[tid] = vacc; __syncthreads();
  for (int st = 128; st > 0; st >>= 1) { if (tid < st) red[tid] += red[tid + st]; __syncthreads(); }
  if (tid == 0) atomicAdd(&varsum[seg], red[0]);
  __syncthreads();
  red[tid] = sacc; __syncthreads();
  for (int st = 128; st > 0; st >>= 1) { if (tid < st) red[tid] += red[tid + st]; __syncthreads(); }
  if (tid == 0) atomicAdd(&seedfg[seg], red[0]);
}

// ---------------------------------------------------------------------------
// Radix sort (descending, LSD, 4x8-bit). Buckets use dd = 255 - digit.
// hist layout: [seg][bin][chunk], 256 chunks of 2048 per segment.
// ---------------------------------------------------------------------------
__global__ __launch_bounds__(256)
void k4_hist(const unsigned* __restrict__ src, unsigned* __restrict__ hist, int shift) {
  __shared__ unsigned lcnt[8][256];
  int tid = threadIdx.x, w = tid >> 5, lane = tid & 31;
  int seg = blockIdx.y;
#pragma unroll
  for (int i = 0; i < 8; ++i) (&lcnt[0][0])[tid + i * 256] = 0u;
  __syncthreads();
  int chunk = blockIdx.x * 8 + w;
  size_t base = (size_t)seg * NPIX + (size_t)chunk * SORTC;
  for (int j = 0; j < 64; ++j) {
    unsigned key = src[base + j * 32 + lane];
    unsigned dd = 255u - ((key >> shift) & 255u);
    atomicAdd(&lcnt[w][dd], 1u);
  }
  __syncthreads();
#pragma unroll
  for (int i = 0; i < 8; ++i) {
    int bin = lane + i * 32;
    hist[(size_t)seg * 65536 + (size_t)bin * NCHUNK + chunk] = lcnt[w][bin];
  }
}

__global__ __launch_bounds__(256)
void k4_scan(unsigned* __restrict__ hist) {
  __shared__ unsigned sums[256];
  int seg = blockIdx.x, tid = threadIdx.x;
  unsigned* h = hist + (size_t)seg * 65536;
  unsigned s = 0;
  for (int i = 0; i < 256; ++i) s += h[tid * 256 + i];
  sums[tid] = s; __syncthreads();
  if (tid == 0) {
    unsigned run = 0;
    for (int i = 0; i < 256; ++i) { unsigned v = sums[i]; sums[i] = run; run += v; }
  }
  __syncthreads();
  unsigned run = sums[tid];
  for (int i = 0; i < 256; ++i) { unsigned v = h[tid * 256 + i]; h[tid * 256 + i] = run; run += v; }
}

__global__ __launch_bounds__(256)
void k4_scatter(const unsigned* __restrict__ src, unsigned* __restrict__ dst,
                const unsigned* __restrict__ hist, int shift) {
  __shared__ volatile unsigned lcnt[8][256];
  int tid = threadIdx.x, w = tid >> 5, lane = tid & 31;
  int seg = blockIdx.y;
#pragma unroll
  for (int i = 0; i < 8; ++i) ((volatile unsigned*)&lcnt[0][0])[tid + i * 256] = 0u;
  __syncthreads();
  int chunk = blockIdx.x * 8 + w;
  size_t base = (size_t)seg * NPIX + (size_t)chunk * SORTC;
  unsigned ltmask = (1u << lane) - 1u;
  for (int j = 0; j < 64; ++j) {
    unsigned key = src[base + j * 32 + lane];
    unsigned dd = 255u - ((key >> shift) & 255u);
    // wave32 stable match-any rank (8 ballots)
    unsigned peers = 0xFFFFFFFFu;
#pragma unroll
    for (int bit = 0; bit < 8; ++bit) {
      unsigned bal = __builtin_amdgcn_ballot_w32((dd >> bit) & 1);
      peers &= ((dd >> bit) & 1) ? bal : ~bal;
    }
    unsigned rank = __popc(peers & ltmask);
    unsigned cntd = __popc(peers);
    unsigned old = lcnt[w][dd];                 // running count before this j
    if (rank == 0) lcnt[w][dd] = old + cntd;    // single writer per digit
    unsigned off = hist[(size_t)seg * 65536 + (size_t)dd * NCHUNK + chunk] + old + rank;
    dst[(size_t)seg * NPIX + off] = key;
  }
}

// ---------------------------------------------------------------------------
// Lovasz: cumulative positive counts + gradient dot product.
// ---------------------------------------------------------------------------
__global__ __launch_bounds__(256)
void k5_lcount(const unsigned* __restrict__ keys, unsigned* __restrict__ lpart) {
  __shared__ unsigned red[256];
  int seg = blockIdx.y, tid = threadIdx.x;
  size_t base = (size_t)seg * NPIX + (size_t)blockIdx.x * LCHUNK;
  unsigned s = 0;
  for (int i = 0; i < 16; ++i) s += keys[base + i * 256 + tid] & 1u;
  red[tid] = s; __syncthreads();
  for (int st = 128; st > 0; st >>= 1) { if (tid < st) red[tid] += red[tid + st]; __syncthreads(); }
  if (tid == 0) lpart[seg * NLCHUNK + blockIdx.x] = red[0];
}

__global__ void k5_scan(unsigned* __restrict__ lpart, unsigned* __restrict__ segG) {
  int seg = blockIdx.x;
  unsigned run = 0;
  for (int i = 0; i < NLCHUNK; ++i) {
    unsigned v = lpart[seg * NLCHUNK + i];
    lpart[seg * NLCHUNK + i] = run;
    run += v;
  }
  segG[seg] = run;
}

__global__ __launch_bounds__(256)
void k5_lovasz(const unsigned* __restrict__ keys, const unsigned* __restrict__ lpart,
               const unsigned* __restrict__ segG, float* __restrict__ instl) {
  __shared__ unsigned ex[256];
  __shared__ float fred[256];
  int seg = blockIdx.y, tid = threadIdx.x;
  size_t base = (size_t)seg * NPIX + (size_t)blockIdx.x * LCHUNK;
  unsigned k[16];
  unsigned cnt = 0;
#pragma unroll
  for (int j = 0; j < 16; ++j) { k[j] = keys[base + tid * 16 + j]; cnt += k[j] & 1u; }
  ex[tid] = cnt; __syncthreads();
  if (tid == 0) {
    unsigned run = 0;
    for (int i = 0; i < 256; ++i) { unsigned v = ex[i]; ex[i] = run; run += v; }
  }
  __syncthreads();
  unsigned c = lpart[seg * NLCHUNK + blockIdx.x] + ex[tid];
  float G = (float)segG[seg];
  int i0 = blockIdx.x * LCHUNK + tid * 16;
  float acc = 0.f;
#pragma unroll
  for (int j = 0; j < 16; ++j) {
    unsigned y = k[j] & 1u;
    c += y;                                      // inclusive positives count
    float err = __uint_as_float(k[j] >> 1);
    float cf = (float)c;
    float i1 = (float)(i0 + j + 1);
    float jac = 1.f - (G - cf) / (G + i1 - cf);
    float grad;
    if (i0 + j > 0) {
      float cp = cf - (float)y;
      float jacp = 1.f - (G - cp) / (G + (i1 - 1.f) - cp);
      grad = jac - jacp;
    } else {
      grad = jac;
    }
    acc += fmaxf(err, 0.f) * grad;
  }
  fred[tid] = acc; __syncthreads();
  for (int st = 128; st > 0; st >>= 1) { if (tid < st) fred[tid] += fred[tid + st]; __syncthreads(); }
  if (tid == 0) atomicAdd(&instl[seg], fred[0]);
}

// ---------------------------------------------------------------------------
__global__ void k6_final(const float* __restrict__ acc, float* __restrict__ out) {
  if (threadIdx.x == 0 && blockIdx.x == 0) {
    float tot = 0.f;
    for (int b = 0; b < NB; ++b) {
      float obj = 0.f, il = 0.f, vl = 0.f, sf = 0.f;
      for (int j = 0; j < NID; ++j) {
        int s = b * NID + j;
        float pres = acc[DER_OFF + s * 8 + 6];
        float cntf = acc[DER_OFF + s * 8 + 7];
        obj += pres;
        il += pres * acc[INSTL_OFF + s];
        vl += pres * (acc[VAR_OFF + s] / (2.f * cntf));      // N_SIGMA=2
        sf += pres * (200.f * acc[SFG_OFF + s]);             // FG_WEIGHT
      }
      float objm = fmaxf(obj, 1.f);
      float inst_loss = il / objm;
      float var_loss = vl / objm;
      float seed_loss = (acc[BST_OFF + b * 3 + 2] + sf) / (float)NPIX;
      float loss_b = 1.f * inst_loss + 10.f * var_loss + 1.f * seed_loss;
      float cls = acc[BST_OFF + b * 3 + 0] / fmaxf(acc[BST_OFF + b * 3 + 1], 1.f);
      tot += loss_b + cls;
    }
    out[0] = tot / (float)NB;
  }
}

// ---------------------------------------------------------------------------
extern "C" void kernel_launch(void* const* d_in, const int* in_sizes, int n_in,
                              void* d_out, int out_size, void* d_ws, size_t ws_size,
                              hipStream_t stream) {
  (void)in_sizes; (void)n_in; (void)out_size; (void)ws_size;
  const float* pred = (const float*)d_in[0];
  const int* inst = (const int*)d_in[1];
  const int* lab = (const int*)d_in[2];
  float* out = (float*)d_out;

  // workspace carve-up (needs ~119 MB)
  unsigned* keysA = (unsigned*)d_ws;
  unsigned* keysB = keysA + (size_t)NSEG * NPIX;
  unsigned* hist = keysB + (size_t)NSEG * NPIX;
  unsigned* lpart = hist + (size_t)NSEG * 65536;
  unsigned* segG = lpart + NSEG * NLCHUNK;
  float* acc = (float*)(segG + NSEG + 4);   // 128B-aligned accumulator block

  k0_zero<<<1, 256, 0, stream>>>(acc);
  k1_stats<<<dim3(64, NB), 256, 0, stream>>>(pred, inst, lab,
                                             acc + STATS_OFF, acc + BST_OFF);
  k2_centers<<<1, 32, 0, stream>>>(acc);
  k3_dist<<<dim3(NLCHUNK, NSEG), 256, 0, stream>>>(pred, inst, acc + DER_OFF,
                                                   keysA, acc + VAR_OFF, acc + SFG_OFF);
  // 4-pass LSD radix sort, descending; even #passes -> result back in keysA
  const unsigned* src[4] = {keysA, keysB, keysA, keysB};
  unsigned* dst[4] = {keysB, keysA, keysB, keysA};
  for (int pass = 0; pass < 4; ++pass) {
    int shift = pass * 8;
    k4_hist<<<dim3(32, NSEG), 256, 0, stream>>>(src[pass], hist, shift);
    k4_scan<<<NSEG, 256, 0, stream>>>(hist);
    k4_scatter<<<dim3(32, NSEG), 256, 0, stream>>>(src[pass], dst[pass], hist, shift);
  }
  k5_lcount<<<dim3(NLCHUNK, NSEG), 256, 0, stream>>>(keysA, lpart);
  k5_scan<<<NSEG, 1, 0, stream>>>(lpart, segG);
  k5_lovasz<<<dim3(NLCHUNK, NSEG), 256, 0, stream>>>(keysA, lpart, segG,
                                                     acc + INSTL_OFF);
  k6_final<<<1, 32, 0, stream>>>(acc, out);
}
